// MoMBlock_57672820851086
// MI455X (gfx1250) — matrix operand order repaired and checked by reading, not compiled, not run on hardware
//
#include <hip/hip_runtime.h>
#include <hip/hip_bf16.h>
#include <math.h>

// ---------------------------------------------------------------------------
// MoM block for MI455X (gfx1250, wave32, WMMA).
// f16 operands staged once; GEMM waves compute 16x48..16x64 tiles with A-frag
// register reuse; in-proj GEMM stages gathered A-tiles in LDS via CDNA5
// async-to-LDS copies. Fragment loads are aligned vector loads (b128).
// ---------------------------------------------------------------------------

typedef __attribute__((ext_vector_type(16))) _Float16 v16h;
typedef __attribute__((ext_vector_type(8)))  _Float16 v8h;
typedef __attribute__((ext_vector_type(8)))  float    v8f;

// Types for the gfx1250 async-copy builtin:
//   void __builtin_amdgcn_global_load_async_to_lds_b128(v4i AS1*, v4i AS3*, imm, imm)
typedef int v4i_vs __attribute__((vector_size(16)));
typedef __attribute__((address_space(1))) v4i_vs* as1_v4i;
typedef __attribute__((address_space(3))) v4i_vs* as3_v4i;

#define DIMC    256
#define NEXP    8
#define LTOK    4096
#define BATCH   4
#define DINNER  512
#define NBE     32           // BATCH * NEXP
#define ROWS_MAX 33280       // 2*B*L + tile padding

#if defined(__has_builtin)
#  if __has_builtin(__builtin_amdgcn_global_load_async_to_lds_b128)
#    define USE_ASYNC_LDS 1
#  endif
#  if __has_builtin(__builtin_amdgcn_s_wait_asynccnt)
#    define WAIT_ASYNC() __builtin_amdgcn_s_wait_asynccnt(0)
#  else
#    define WAIT_ASYNC() asm volatile("s_wait_asynccnt 0x0" ::: "memory")
#  endif
#endif

#define WMMA_F16(a, b, c) \
  __builtin_amdgcn_wmma_f32_16x16x32_f16(false, (a), false, (b), (short)0, (c), false, false)

// A-fragment (16x32 f16): lane m (0-15) K={0..7,16..23}, lane m+16 K={8..15,24..31}
// Two aligned 16B chunks concatenated -> two b128 loads, no element moves.
__device__ inline v16h fragA_h(const _Float16* __restrict__ rowp, int lane) {
  const int kh = (lane & 16) ? 8 : 0;
  const v8h lo = *(const v8h*)(rowp + kh);
  const v8h hi = *(const v8h*)(rowp + kh + 16);
  return __builtin_shufflevector(lo, hi, 0, 1, 2, 3, 4, 5, 6, 7,
                                 8, 9, 10, 11, 12, 13, 14, 15);
}
// B-fragment (32x16 f16): lane n (0-15) K={0..7,16..23}, lane n+16 K={8..15,24..31}
// Two aligned 16B chunks concatenated -> two b128 loads.
__device__ inline v16h fragB_h(const _Float16* __restrict__ rowp, int lane) {
  const int kh = (lane & 16) ? 8 : 0;
  const v8h lo = *(const v8h*)(rowp + kh);
  const v8h hi = *(const v8h*)(rowp + kh + 16);
  return __builtin_shufflevector(lo, hi, 0, 1, 2, 3, 4, 5, 6, 7,
                                 8, 9, 10, 11, 12, 13, 14, 15);
}

// ---------------- f32 -> f16 weight conversion ----------------
__global__ void cvt_kernel(const float* __restrict__ in, _Float16* __restrict__ out, long n) {
  const long i = (long)blockIdx.x * 256 + threadIdx.x;
  if (i < n) out[i] = (_Float16)in[i];
}

// ---------------- LayerNorm + router (1 wave per token) ----------------
__global__ void ln_router_kernel(const float* __restrict__ x,
                                 const float* __restrict__ ln_g,
                                 const float* __restrict__ ln_b,
                                 const float* __restrict__ Wg,
                                 _Float16* __restrict__ xnh,
                                 float* __restrict__ gate,
                                 unsigned char* __restrict__ mask,
                                 float* __restrict__ fsum,
                                 float* __restrict__ Psum) {
  const int wid = threadIdx.x >> 5, lane = threadIdx.x & 31;
  const int tok = blockIdx.x * 8 + wid;           // 16384 tokens
  const int b = tok >> 12, l = tok & 4095;
  float v[8];
  float s = 0.f;
#pragma unroll
  for (int i = 0; i < 8; ++i) {
    const int c = i * 32 + lane;
    v[i] = x[((long)(b * DIMC + c)) * LTOK + l];
    s += v[i];
  }
#pragma unroll
  for (int o = 16; o > 0; o >>= 1) s += __shfl_xor(s, o);
  const float mu = s * (1.0f / 256.0f);
  float vs = 0.f;
#pragma unroll
  for (int i = 0; i < 8; ++i) { const float d = v[i] - mu; vs += d * d; }
#pragma unroll
  for (int o = 16; o > 0; o >>= 1) vs += __shfl_xor(vs, o);
  const float rstd = rsqrtf(vs * (1.0f / 256.0f) + 1e-5f);

  float p[NEXP];
#pragma unroll
  for (int e = 0; e < NEXP; ++e) p[e] = 0.f;
#pragma unroll
  for (int i = 0; i < 8; ++i) {
    const int c = i * 32 + lane;
    const float xv = (v[i] - mu) * rstd * ln_g[c] + ln_b[c];
    xnh[((long)tok) * DIMC + c] = (_Float16)xv;
#pragma unroll
    for (int e = 0; e < NEXP; ++e) p[e] += xv * Wg[c * NEXP + e];
  }
#pragma unroll
  for (int e = 0; e < NEXP; ++e) {
#pragma unroll
    for (int o = 16; o > 0; o >>= 1) p[e] += __shfl_xor(p[e], o);
  }
  if (lane == 0) {
    float mx = p[0];
    for (int e = 1; e < NEXP; ++e) mx = fmaxf(mx, p[e]);
    float den = 0.f;
    for (int e = 0; e < NEXP; ++e) { p[e] = __expf(p[e] - mx); den += p[e]; }
    const float inv = 1.0f / den;
    int e1 = 0; float b1 = -1.f;
    for (int e = 0; e < NEXP; ++e) { p[e] *= inv; if (p[e] > b1) { b1 = p[e]; e1 = e; } }
    int e2 = 0; float b2 = -1.f;
    for (int e = 0; e < NEXP; ++e) { if (e != e1 && p[e] > b2) { b2 = p[e]; e2 = e; } }
    for (int e = 0; e < NEXP; ++e) atomicAdd(&Psum[e], p[e]);
    gate[((long)(b * NEXP + e1)) * LTOK + l] = b1;
    gate[((long)(b * NEXP + e2)) * LTOK + l] = b2;
    mask[(b * NEXP + e1) * LTOK + l] = 1;
    mask[(b * NEXP + e2) * LTOK + l] = 1;
    atomicAdd(&fsum[e1], 1.0f);
    atomicAdd(&fsum[e2], 1.0f);
  }
}

// ---------------- routed-token counts / offsets / stable pack ----------------
__global__ void count_kernel(const unsigned char* __restrict__ mask, int* __restrict__ nArr) {
  const int be = blockIdx.x, lane = threadIdx.x;
  int cnt = 0;
  for (int t0 = 0; t0 < LTOK; t0 += 32) {
    const int m = mask[be * LTOK + t0 + lane];
    cnt += __popc((unsigned)__ballot(m != 0));
  }
  if (lane == 0) nArr[be] = cnt;
}

__global__ void offsets_kernel(const int* __restrict__ nArr, int* __restrict__ npArr,
                               int* __restrict__ offArr) {
  if (blockIdx.x == 0 && threadIdx.x == 0) {
    int acc = 0;
    for (int i = 0; i < NBE; ++i) {
      const int n = nArr[i];
      const int np = (n + 15) & ~15;
      npArr[i] = np; offArr[i] = acc; acc += np;
    }
  }
}

__global__ void pack_kernel(const unsigned char* __restrict__ mask,
                            const int* __restrict__ npArr, const int* __restrict__ offArr,
                            int* __restrict__ idx) {
  const int be = blockIdx.x, lane = threadIdx.x;
  const int off = offArr[be];
  int cnt = 0;
  for (int t0 = 0; t0 < LTOK; t0 += 32) {
    const int t = t0 + lane;
    const int m = mask[be * LTOK + t];
    const unsigned mm = (unsigned)__ballot(m != 0);
    const int pos = cnt + __popc(mm & ((1u << lane) - 1u));
    if (m) idx[off + pos] = t;
    cnt += __popc(mm);
  }
  const int np = npArr[be];
  for (int j = cnt + lane; j < np; j += 32) idx[off + j] = -1;
}

// ---------------- GEMM1: xz = gather(xn) @ W_in^T  (K=256, N=1024) --------------
// Block = 8 waves; A tile (16 gathered rows) staged in LDS (async copy) and
// shared by all waves; each wave computes a 16x64 slab (4 WMMA accumulators).
__global__ void __launch_bounds__(256) gemm_in_kernel(
    const _Float16* __restrict__ xnh, const _Float16* __restrict__ W_in_h,
    const int* __restrict__ idx, const int* __restrict__ npArr,
    const int* __restrict__ offArr, float* __restrict__ xz) {
  __shared__ __align__(64) _Float16 sA[16 * 256];
  const int job = blockIdx.x;               // NBE * 256 * 2 jobs
  const int be = job >> 9;
  const int rem = job & 511;
  const int mt = rem >> 1, nh = rem & 1;
  const int np = npArr[be];
  if (mt * 16 >= np) return;                // block-uniform early exit
  const int off = offArr[be];
  const int b = be >> 3, e = be & 7;

  // ---- stage A tile: 16 rows x 256 K of f16 (8 KB), 16B chunk per thread x2
#pragma unroll
  for (int s = 0; s < 2; ++s) {
    const int chunk = threadIdx.x + s * 256;   // 0..511
    const int row = chunk >> 5, ck = chunk & 31;
    const int l = idx[off + mt * 16 + row];
    _Float16* lp = &sA[row * 256 + ck * 8];
    if (l >= 0) {
      const _Float16* gp = xnh + ((long)(b * LTOK + l)) * 256 + ck * 8;
#ifdef USE_ASYNC_LDS
      __builtin_amdgcn_global_load_async_to_lds_b128(
          (as1_v4i)(unsigned long long)gp, (as3_v4i)lp, 0, 0);
#else
      *(v4i_vs*)lp = *(const v4i_vs*)gp;
#endif
    } else {
      const v4i_vs z = {0, 0, 0, 0};
      *(v4i_vs*)lp = z;
    }
  }
#ifdef USE_ASYNC_LDS
  WAIT_ASYNC();
#endif
  __syncthreads();

  const int wid = threadIdx.x >> 5, lane = threadIdx.x & 31;
  const int colbase = nh * 512 + wid * 64;
  const _Float16* wb = W_in_h + ((long)(e * 1024 + colbase + (lane & 15))) * 256;
  const _Float16* arow = &sA[(lane & 15) * 256];
  v8f a0 = {}, a1 = {}, a2 = {}, a3 = {};
#pragma unroll
  for (int k0 = 0; k0 < 256; k0 += 32) {
    const v16h a = fragA_h(arow + k0, lane);
    a0 = WMMA_F16(a, fragB_h(wb + 0 * 16 * 256 + k0, lane), a0);
    a1 = WMMA_F16(a, fragB_h(wb + 1 * 16 * 256 + k0, lane), a1);
    a2 = WMMA_F16(a, fragB_h(wb + 2 * 16 * 256 + k0, lane), a2);
    a3 = WMMA_F16(a, fragB_h(wb + 3 * 16 * 256 + k0, lane), a3);
  }
  const int col0 = colbase + (lane & 15);
  const long crow = off + mt * 16 + ((lane & 16) ? 8 : 0);
#pragma unroll
  for (int r = 0; r < 8; ++r) {
    float* cp = xz + (crow + r) * 1024 + col0;
    cp[0] = a0[r]; cp[16] = a1[r]; cp[32] = a2[r]; cp[48] = a3[r];
  }
}

// ---------------- causal depthwise conv (d_conv=4) + SiLU -> f16 ----------------
__global__ void conv_silu_kernel(const float* __restrict__ xz, const float* __restrict__ conv_w,
                                 const float* __restrict__ conv_b,
                                 const int* __restrict__ nArr, const int* __restrict__ npArr,
                                 const int* __restrict__ offArr, _Float16* __restrict__ xi_h) {
  const int be = blockIdx.y, e = be & 7;
  const long flat = (long)blockIdx.x * blockDim.x + threadIdx.x;
  const int t = (int)(flat >> 9), c = (int)(flat & 511);
  const int np = npArr[be];
  if (t >= np) return;
  const int off = offArr[be];
  const long row = off + t;
  if (t >= nArr[be]) { xi_h[row * 512 + c] = (_Float16)0.f; return; }
  float acc = conv_b[e * 512 + c];
#pragma unroll
  for (int j = 0; j < 4; ++j) {
    const int ts = t + j - 3;
    const float xv = (ts >= 0) ? xz[(long)(off + ts) * 1024 + c] : 0.f;
    acc = fmaf(xv, conv_w[(e * 512 + c) * 4 + j], acc);
  }
  xi_h[row * 512 + c] = (_Float16)(acc / (1.0f + __expf(-acc)));
}

// ---------------- GEMM2: dbc = xi @ W_x^T (K=512, N=48), 16x48/wave ----------------
__global__ void __launch_bounds__(256) gemm_dbc_kernel(
    const _Float16* __restrict__ xi_h, const _Float16* __restrict__ W_x_h,
    const int* __restrict__ npArr, const int* __restrict__ offArr,
    float* __restrict__ dbc) {
  const int wid = threadIdx.x >> 5, lane = threadIdx.x & 31;
  const int job = blockIdx.x * 8 + wid;        // NBE * 256 wave-jobs
  const int be = job >> 8;
  if (be >= NBE) return;
  const int mt = job & 255;
  const int np = npArr[be];
  if (mt * 16 >= np) return;
  const int off = offArr[be];
  const int e = be & 7;
  const _Float16* abase = xi_h + ((long)(off + mt * 16 + (lane & 15))) * 512;
  const _Float16* wb = W_x_h + ((long)(e * 48 + (lane & 15))) * 512;
  v8f a0 = {}, a1 = {}, a2 = {};
#pragma unroll 2
  for (int k0 = 0; k0 < 512; k0 += 32) {
    const v16h a = fragA_h(abase + k0, lane);
    a0 = WMMA_F16(a, fragB_h(wb + 0 * 16 * 512 + k0, lane), a0);
    a1 = WMMA_F16(a, fragB_h(wb + 1 * 16 * 512 + k0, lane), a1);
    a2 = WMMA_F16(a, fragB_h(wb + 2 * 16 * 512 + k0, lane), a2);
  }
  const int col = lane & 15;
  const long crow = off + mt * 16 + ((lane & 16) ? 8 : 0);
#pragma unroll
  for (int r = 0; r < 8; ++r) {
    float* cp = dbc + (crow + r) * 48 + col;
    cp[0] = a0[r]; cp[16] = a1[r]; cp[32] = a2[r];
  }
}

// ---------------- sequential SSM scan (1 WG per (b,e), 1 lane per channel) ----------------
__global__ void __launch_bounds__(512) scan_kernel(
    const float* __restrict__ dbc, const _Float16* __restrict__ xi_h,
    const float* __restrict__ xz, const float* __restrict__ W_dt,
    const float* __restrict__ dt_bias, const float* __restrict__ A_log,
    const float* __restrict__ Dp, const int* __restrict__ nArr,
    const int* __restrict__ offArr, _Float16* __restrict__ yg_h) {
  const int be = blockIdx.x, e = be & 7;
  const int c = threadIdx.x;
  const int n = nArr[be], off = offArr[be];
  float Wd[16], Av[16], h[16];
#pragma unroll
  for (int k = 0; k < 16; ++k) {
    Wd[k] = W_dt[((long)(e * 512 + c)) * 16 + k];
    Av[k] = -__expf(A_log[((long)(e * 512 + c)) * 16 + k]);
    h[k] = 0.f;
  }
  const float dtb = dt_bias[e * 512 + c];
  const float Dc  = Dp[e * 512 + c];
  for (int t = 0; t < n; ++t) {
    const long row = off + t;
    const float* dr = dbc + row * 48;
    if (c == 0) __builtin_prefetch(dbc + (row + 8) * 48, 0, 1);  // global_prefetch
    float dv = dtb;
#pragma unroll
    for (int k = 0; k < 16; ++k) dv = fmaf(dr[k], Wd[k], dv);
    const float dt = (dv > 20.f) ? dv : log1pf(__expf(dv));      // softplus
    const float xv = (float)xi_h[row * 512 + c];
    const float zv = xz[row * 1024 + 512 + c];
    const float dtx = dt * xv;
    float y = 0.f;
#pragma unroll
    for (int s = 0; s < 16; ++s) {
      const float dA = __expf(dt * Av[s]);
      h[s] = fmaf(dA, h[s], dtx * dr[16 + s]);
      y = fmaf(h[s], dr[32 + s], y);
    }
    y = fmaf(Dc, xv, y);
    y *= zv / (1.0f + __expf(-zv));                              // * silu(z)
    yg_h[row * 512 + c] = (_Float16)y;
  }
}

// ---------------- GEMM3: yg @ W_out^T (K=512, N=256), 16x64/wave, gated scatter ----
__global__ void __launch_bounds__(256) gemm_out_kernel(
    const _Float16* __restrict__ yg_h, const _Float16* __restrict__ W_out_h,
    const int* __restrict__ idx, const int* __restrict__ npArr,
    const int* __restrict__ offArr, const float* __restrict__ gate,
    float* __restrict__ xm) {
  const int wid = threadIdx.x >> 5, lane = threadIdx.x & 31;
  const int job = blockIdx.x * 8 + wid;        // NBE * 256 * 4 wave-jobs
  const int be = job >> 10;
  if (be >= NBE) return;
  const int rem = job & 1023;
  const int mt = rem >> 2, nq = rem & 3;
  const int np = npArr[be];
  if (mt * 16 >= np) return;
  const int off = offArr[be];
  const int b = be >> 3, e = be & 7;
  const _Float16* abase = yg_h + ((long)(off + mt * 16 + (lane & 15))) * 512;
  const _Float16* wb = W_out_h + ((long)(e * 256 + nq * 64 + (lane & 15))) * 512;
  v8f a0 = {}, a1 = {}, a2 = {}, a3 = {};
#pragma unroll 2
  for (int k0 = 0; k0 < 512; k0 += 32) {
    const v16h a = fragA_h(abase + k0, lane);
    a0 = WMMA_F16(a, fragB_h(wb + 0 * 16 * 512 + k0, lane), a0);
    a1 = WMMA_F16(a, fragB_h(wb + 1 * 16 * 512 + k0, lane), a1);
    a2 = WMMA_F16(a, fragB_h(wb + 2 * 16 * 512 + k0, lane), a2);
    a3 = WMMA_F16(a, fragB_h(wb + 3 * 16 * 512 + k0, lane), a3);
  }
  const int col0 = nq * 64 + (lane & 15);
  const int r0 = (lane & 16) ? 8 : 0;
#pragma unroll
  for (int r = 0; r < 8; ++r) {
    const int m = mt * 16 + r0 + r;
    const int l = idx[off + m];
    if (l >= 0) {
      const float g = gate[((long)(b * NEXP + e)) * LTOK + l];
      float* xp = xm + ((long)(b * LTOK + l)) * DIMC + col0;
      atomicAdd(xp +  0, g * a0[r]);
      atomicAdd(xp + 16, g * a1[r]);
      atomicAdd(xp + 32, g * a2[r]);
      atomicAdd(xp + 48, g * a3[r]);
    }
  }
}

// ---------------- residual + transpose back + aux loss ----------------
__global__ void final_kernel(const float* __restrict__ x, const float* __restrict__ gamma,
                             const float* __restrict__ xm, const float* __restrict__ fsum,
                             const float* __restrict__ Psum, float* __restrict__ out) {
  const long TOT = (long)BATCH * DIMC * LTOK;
  const long i = (long)blockIdx.x * blockDim.x + threadIdx.x;
  if (i < TOT) {
    const int b = (int)(i / ((long)DIMC * LTOK));
    const int rem = (int)(i % ((long)DIMC * LTOK));
    const int c = rem / LTOK, l = rem % LTOK;
    out[i] = x[i] + gamma[c] * xm[((long)(b * LTOK + l)) * DIMC + c];
  }
  if (i == 0) {
    float aux = 0.f;
    const float invBL = 1.0f / (float)(BATCH * LTOK);
    for (int e = 0; e < NEXP; ++e)
      aux += (fsum[e] * invBL * 0.5f) * (Psum[e] * invBL);
    out[TOT] = (float)NEXP * aux;
  }
}

extern "C" void kernel_launch(void* const* d_in, const int* in_sizes, int n_in,
                              void* d_out, int out_size, void* d_ws, size_t ws_size,
                              hipStream_t stream) {
  (void)in_sizes; (void)n_in; (void)out_size; (void)ws_size;
  const float* x       = (const float*)d_in[0];
  const float* ln_g    = (const float*)d_in[1];
  const float* ln_b    = (const float*)d_in[2];
  const float* gamma   = (const float*)d_in[3];
  const float* Wg      = (const float*)d_in[4];
  const float* W_in    = (const float*)d_in[5];
  const float* conv_w  = (const float*)d_in[6];
  const float* conv_b  = (const float*)d_in[7];
  const float* W_x     = (const float*)d_in[8];
  const float* W_dt    = (const float*)d_in[9];
  const float* dt_bias = (const float*)d_in[10];
  const float* A_log   = (const float*)d_in[11];
  const float* Dp      = (const float*)d_in[12];
  const float* W_out   = (const float*)d_in[13];
  float* out = (float*)d_out;

  const long TOKENS = (long)BATCH * LTOK;   // 16384

  // 256B-aligned workspace carving
  char* p = (char*)d_ws;
  auto alloc = [&](size_t bytes) -> void* {
    void* r = (void*)p; p += (bytes + 255) & ~(size_t)255; return r;
  };
  _Float16* xnh    = (_Float16*)alloc(TOKENS * DIMC * 2);
  float*    gate   = (float*)   alloc(((long)NBE * LTOK + 16) * 4);  // + fsum/Psum
  float*    fsum   = gate + (long)NBE * LTOK;
  float*    Psum   = fsum + 8;
  float*    xm     = (float*)   alloc(TOKENS * DIMC * 4);
  float*    xz     = (float*)   alloc((long)ROWS_MAX * 1024 * 4);
  _Float16* xi_h   = (_Float16*)alloc((long)ROWS_MAX * 512 * 2);
  float*    dbc    = (float*)   alloc((long)ROWS_MAX * 48 * 4);
  _Float16* yg_h   = (_Float16*)alloc((long)ROWS_MAX * 512 * 2);
  _Float16* W_in_h = (_Float16*)alloc((long)NEXP * 1024 * 256 * 2);
  _Float16* W_x_h  = (_Float16*)alloc((long)NEXP * 48 * 512 * 2);
  _Float16* W_out_h= (_Float16*)alloc((long)NEXP * 256 * 512 * 2);
  int*      nArr   = (int*)     alloc(NBE * 4);
  int*      npArr  = (int*)     alloc(NBE * 4);
  int*      offArr = (int*)     alloc(NBE * 4);
  int*      idx    = (int*)     alloc((long)ROWS_MAX * 4);
  unsigned char* mask = (unsigned char*)alloc((long)NBE * LTOK);

  (void)hipMemsetAsync(gate, 0, (size_t)((long)NBE * LTOK + 16) * 4, stream);
  (void)hipMemsetAsync(xm, 0, (size_t)(TOKENS * DIMC) * 4, stream);
  (void)hipMemsetAsync(mask, 0, (size_t)NBE * LTOK, stream);

  // 0. weight conversion to f16 (one pass, L2-resident afterwards)
  {
    const long nw1 = (long)NEXP * 1024 * 256;
    const long nw2 = (long)NEXP * 48 * 512;
    const long nw3 = (long)NEXP * 256 * 512;
    cvt_kernel<<<(int)((nw1 + 255) / 256), 256, 0, stream>>>(W_in, W_in_h, nw1);
    cvt_kernel<<<(int)((nw2 + 255) / 256), 256, 0, stream>>>(W_x, W_x_h, nw2);
    cvt_kernel<<<(int)((nw3 + 255) / 256), 256, 0, stream>>>(W_out, W_out_h, nw3);
  }
  // 1. LN + router (writes f16 activations directly)
  ln_router_kernel<<<TOKENS / 8, 256, 0, stream>>>(x, ln_g, ln_b, Wg, xnh, gate, mask, fsum, Psum);
  // 2. counts -> padded offsets -> packed index lists
  count_kernel<<<NBE, 32, 0, stream>>>(mask, nArr);
  offsets_kernel<<<1, 32, 0, stream>>>(nArr, npArr, offArr);
  pack_kernel<<<NBE, 32, 0, stream>>>(mask, npArr, offArr, idx);
  // 3. in-projection GEMM, LDS-staged A tile (async copy), 16x64 per wave
  gemm_in_kernel<<<NBE * 256 * 2, 256, 0, stream>>>(xnh, W_in_h, idx, npArr, offArr, xz);
  // 4. causal conv + SiLU -> f16
  {
    dim3 g((LTOK * DINNER) / 256, NBE, 1);
    conv_silu_kernel<<<g, 256, 0, stream>>>(xz, conv_w, conv_b, nArr, npArr, offArr, xi_h);
  }
  // 5. x-projection GEMM (dt/B/C), 16x48 per wave
  gemm_dbc_kernel<<<(NBE * 256) / 8, 256, 0, stream>>>(xi_h, W_x_h, npArr, offArr, dbc);
  // 6. sequential selective scan (prefetch ahead)
  scan_kernel<<<NBE, 512, 0, stream>>>(dbc, xi_h, xz, W_dt, dt_bias, A_log, Dp, nArr, offArr, yg_h);
  // 7. out-projection GEMM + gated scatter-add, 16x64 per wave
  gemm_out_kernel<<<(NBE * 256 * 4) / 8, 256, 0, stream>>>(yg_h, W_out_h, idx, npArr, offArr, gate, xm);
  // 8. residual + transpose + aux loss
  final_kernel<<<(int)((TOKENS * DIMC + 255) / 256), 256, 0, stream>>>(x, gamma, xm, fsum, Psum, out);
}
